// IoTGNN_81088982548949
// MI455X (gfx1250) — matrix operand order repaired and checked by reading, not compiled, not run on hardware
//
#include <hip/hip_runtime.h>
#include <hip/hip_bf16.h>

typedef __attribute__((ext_vector_type(16))) _Float16 v16h;
typedef __attribute__((ext_vector_type(8)))  float    v8f;

// ---------------------------------------------------------------- degrees ---
__global__ void k_deg_init(float* deg, int N) {
  int i = blockIdx.x * blockDim.x + threadIdx.x;
  if (i < N) deg[i] = 1.0f;                       // self-loop contribution
}

__global__ void k_deg_accum(const long long* __restrict__ ei, float* deg, int E) {
  int e = blockIdx.x * blockDim.x + threadIdx.x;
  if (e < E) atomicAdd(&deg[(int)ei[E + e]], 1.0f);   // dst row of edge_index
}

__global__ void k_rsqrt_inplace(float* d, int N) {
  int i = blockIdx.x * blockDim.x + threadIdx.x;
  if (i < N) d[i] = rsqrtf(d[i]);
}

// ------------------------------------------------------- WMMA f16 GEMM ------
// H[N x F] = X[N x K] @ W[K x F]  (row-major, f32 in/out, f16 matrix core,
// f32 accumulate), fused epilogue:  A0[i][f] = dinv[i]^2 * H[i][f] + bias[f]
// (GCN self-loop term + bias, full overwrite of the aggregation buffer).
//
// W is staged once per block in LDS, transposed to [F][K+2] f16 (pad breaks
// the stride-256B bank conflict). Each wave holds its 16xK A strip in
// registers (f16) and sweeps all F/16 column tiles: 4 unrolled v_wmma each.
template <int K>
__global__ void __launch_bounds__(256)
k_gemm_wmma(const float* __restrict__ X, const float* __restrict__ W,
            const float* __restrict__ dinv, const float* __restrict__ bias,
            float* __restrict__ H, float* __restrict__ A0, int N, int F) {
  constexpr int KP = K + 2;                       // padded LDS row stride
  __shared__ _Float16 Bs[128 * KP];               // <= 128 cols * 130 * 2B = 33.3 KB

  const int tid = threadIdx.x;
  for (int idx = tid; idx < K * F; idx += blockDim.x) {
    int k = idx / F, f = idx - k * F;             // coalesced global read
    Bs[f * KP + k] = (_Float16)W[idx];            // transposed LDS write
  }
  __syncthreads();

  const int wave  = tid >> 5;
  const int lane  = tid & 31;
  const int tileM = blockIdx.x * (blockDim.x >> 5) + wave;
  if (tileM >= (N >> 4)) return;                  // wave-uniform; after barrier

  const int  m0 = tileM << 4;
  const int  l  = lane & 15;
  const bool hi = lane >= 16;
  const int  offA = hi ? 8 : 0;                   // ISA 16-bit A 16x32 layout
  const int  offB = hi ? 8 : 0;                   // ISA 16-bit B 32x16 layout
  const int  rbase = hi ? 8 : 0;                  // ISA 32-bit C/D 16x16 layout

  // Load this wave's full 16 x K A strip once (f32 -> f16).
  const float* Arow = X + (size_t)(m0 + l) * K;
  v16h a[K / 32];
#pragma unroll
  for (int kbi = 0; kbi < K / 32; ++kbi) {
#pragma unroll
    for (int v = 0; v < 8; ++v) {
      const int ka = kbi * 32 + offA + ((v < 4) ? 2 * v : 2 * v + 8);
      a[kbi][2 * v]     = (_Float16)Arow[ka];
      a[kbi][2 * v + 1] = (_Float16)Arow[ka + 1];
    }
  }

  for (int n0 = 0; n0 < F; n0 += 16) {
    const _Float16* bcol = Bs + (n0 + l) * KP;    // this lane's B column
    v8f c = {0.f, 0.f, 0.f, 0.f, 0.f, 0.f, 0.f, 0.f};
#pragma unroll
    for (int kbi = 0; kbi < K / 32; ++kbi) {
      v16h b;
#pragma unroll
      for (int v = 0; v < 8; ++v) {
        const int kk = kbi * 32 + offB + ((v < 4) ? 2 * v : 2 * v + 8);   // even -> 4B-aligned pair
        b[2 * v]     = bcol[kk];
        b[2 * v + 1] = bcol[kk + 1];
      }
      c = __builtin_amdgcn_wmma_f32_16x16x32_f16(
          /*neg_a=*/false, a[kbi], /*neg_b=*/false, b,
          /*c_mod=*/(short)0, c, /*reuse_a=*/false, /*reuse_b=*/false);
    }
    const float bcst = bias[n0 + l];
#pragma unroll
    for (int r = 0; r < 8; ++r) {
      const int row = m0 + rbase + r;
      const size_t idx = (size_t)row * F + n0 + l;
      const float h = c[r];
      const float di = dinv[row];
      H[idx]  = h;                                // message source for scatter
      A0[idx] = di * di * h + bcst;               // self-loop + bias init
    }
  }
}

// --------------------------------------------------- GCN edge scatter -------
// out[dst] += dinv[src]*dinv[dst] * h[src]; F/4 lanes per edge, float4 gather.
__global__ void k_conv_scatter(const long long* __restrict__ ei,
                               const float* __restrict__ h,
                               const float* __restrict__ dinv,
                               float* out, int E, int F) {
  const int lpe = F >> 2;                         // lanes per edge
  long long t = (long long)blockIdx.x * blockDim.x + threadIdx.x;
  long long e = t / lpe;
  int lane = (int)(t - e * lpe);
  if (e >= E) return;
  int s = (int)ei[e];
  int d = (int)ei[E + e];
  float w = dinv[s] * dinv[d];
  const float4 v = *(const float4*)(h + (size_t)s * F + lane * 4);
  float* o = out + (size_t)d * F + lane * 4;
  atomicAdd(o + 0, w * v.x);
  atomicAdd(o + 1, w * v.y);
  atomicAdd(o + 2, w * v.z);
  atomicAdd(o + 3, w * v.w);
}

// ------------------------------------------------------------- batchnorm ----
__global__ void k_zero(float* p, int n) {
  int i = blockIdx.x * blockDim.x + threadIdx.x;
  if (i < n) p[i] = 0.f;
}

// blockDim.x == F; each thread owns one column over a row chunk (coalesced).
__global__ void k_bn_stats(const float* __restrict__ x, float* sum, float* sumsq,
                           int N, int F, int rowsPerBlock) {
  int f  = threadIdx.x;
  int r0 = blockIdx.x * rowsPerBlock;
  int r1 = r0 + rowsPerBlock; if (r1 > N) r1 = N;
  float s = 0.f, q = 0.f;
  for (int r = r0; r < r1; ++r) {
    float v = x[(size_t)r * F + f];
    s += v; q += v * v;
  }
  atomicAdd(&sum[f], s);
  atomicAdd(&sumsq[f], q);
}

__global__ void k_bn_finalize(const float* sum, const float* sumsq,
                              const float* g, const float* be,
                              float* scale, float* shift, int N, int F) {
  int f = blockIdx.x * blockDim.x + threadIdx.x;
  if (f >= F) return;
  float inv = 1.0f / (float)N;
  float mu  = sum[f] * inv;
  float var = sumsq[f] * inv - mu * mu;           // biased variance (torch BN)
  float sc  = g[f] * rsqrtf(var + 1e-5f);
  scale[f] = sc;
  shift[f] = be[f] - mu * sc;
}

__global__ void k_bn_apply_relu(float* x, const float* __restrict__ scale,
                                const float* __restrict__ shift, int N, int F) {
  int t = blockIdx.x * blockDim.x + threadIdx.x;
  if (t >= N * F) return;
  int f = t % F;
  float v = scale[f] * x[t] + shift[f];
  x[t] = v > 0.f ? v : 0.f;
}

// ------------------------------------------------------------- final head ---
__global__ void k_final(const float* __restrict__ a, const float* __restrict__ Wf,
                        const float* __restrict__ bf, float* out, int N, int F) {
  int i = blockIdx.x * blockDim.x + threadIdx.x;
  if (i >= N) return;
  const float* row = a + (size_t)i * F;
  float o0 = bf[0], o1 = bf[1];
  for (int k = 0; k < F; ++k) {
    float v = row[k];
    o0 += v * Wf[k * 2 + 0];
    o1 += v * Wf[k * 2 + 1];
  }
  out[i * 2 + 0] = o0;
  out[i * 2 + 1] = o1;
}

// ----------------------------------------------------------------------------
extern "C" void kernel_launch(void* const* d_in, const int* in_sizes, int n_in,
                              void* d_out, int out_size, void* d_ws, size_t ws_size,
                              hipStream_t stream) {
  const float*     x   = (const float*)d_in[0];
  const long long* ei  = (const long long*)d_in[1];   // int64 [2, E] row-major
  const float*     W1  = (const float*)d_in[2];
  const float*     b1  = (const float*)d_in[3];
  const float*     g1  = (const float*)d_in[4];
  const float*     be1 = (const float*)d_in[5];
  const float*     W2  = (const float*)d_in[6];
  const float*     b2  = (const float*)d_in[7];
  const float*     g2  = (const float*)d_in[8];
  const float*     be2 = (const float*)d_in[9];
  const float*     Wf  = (const float*)d_in[10];
  const float*     bf  = (const float*)d_in[11];
  float*           out = (float*)d_out;

  const int Din = 128, H = 128, H2 = 64;
  const int N = in_sizes[0] / Din;
  const int E = in_sizes[1] / 2;

  float* ws    = (float*)d_ws;
  float* dinv  = ws;                                  // N
  float* hbuf  = dinv + N;                            // N*H
  float* abuf  = hbuf + (size_t)N * H;                // N*H
  float* stats = abuf + (size_t)N * H;                // 4*H
  float* s_sum = stats, *s_sq = stats + H, *s_sc = stats + 2 * H, *s_sh = stats + 3 * H;

  auto cdiv = [](long long a, long long b) { return (unsigned)((a + b - 1) / b); };

  // --- degrees -> dinv (in place) ---
  k_deg_init     <<<cdiv(N, 256), 256, 0, stream>>>(dinv, N);
  k_deg_accum    <<<cdiv(E, 256), 256, 0, stream>>>(ei, dinv, E);
  k_rsqrt_inplace<<<cdiv(N, 256), 256, 0, stream>>>(dinv, N);

  const int statBlocks = 512;
  const int rpb = (N + statBlocks - 1) / statBlocks;
  const unsigned gemmGrid = cdiv(N / 16, 8);          // 8 row-tiles per block

  // --- layer 1: conv(x, W1) -> BN -> ReLU ---
  k_gemm_wmma<128><<<gemmGrid, 256, 0, stream>>>(x, W1, dinv, b1, hbuf, abuf, N, H);
  k_conv_scatter<<<cdiv((long long)E * (H / 4), 256), 256, 0, stream>>>(ei, hbuf, dinv, abuf, E, H);
  k_zero<<<1, 256, 0, stream>>>(s_sum, 2 * H);
  k_bn_stats<<<statBlocks, H, 0, stream>>>(abuf, s_sum, s_sq, N, H, rpb);
  k_bn_finalize<<<1, H, 0, stream>>>(s_sum, s_sq, g1, be1, s_sc, s_sh, N, H);
  k_bn_apply_relu<<<cdiv((long long)N * H, 256), 256, 0, stream>>>(abuf, s_sc, s_sh, N, H);

  // --- layer 2: conv(a1, W2) -> BN -> ReLU ---
  k_gemm_wmma<128><<<gemmGrid, 256, 0, stream>>>(abuf, W2, dinv, b2, hbuf, abuf, N, H2);
  k_conv_scatter<<<cdiv((long long)E * (H2 / 4), 256), 256, 0, stream>>>(ei, hbuf, dinv, abuf, E, H2);
  k_zero<<<1, 256, 0, stream>>>(s_sum, 2 * H);
  k_bn_stats<<<statBlocks, H2, 0, stream>>>(abuf, s_sum, s_sq, N, H2, rpb);
  k_bn_finalize<<<1, H2, 0, stream>>>(s_sum, s_sq, g2, be2, s_sc, s_sh, N, H2);
  k_bn_apply_relu<<<cdiv((long long)N * H2, 256), 256, 0, stream>>>(abuf, s_sc, s_sh, N, H2);

  // --- head ---
  k_final<<<cdiv(N, 256), 256, 0, stream>>>(abuf, Wf, bf, out, N, H2);
}
